// MultiHeadAttention_37366215475515
// MI455X (gfx1250) — matrix operand order repaired and checked by reading, not compiled, not run on hardware
//
#include <hip/hip_runtime.h>
#include <hip/hip_bf16.h>

// ---------------------------------------------------------------------------
// MHA for MI455X (gfx1250, wave32). bf16 WMMA (v_wmma_f32_16x16x32_bf16),
// fp32 accumulate, flash-attention streaming softmax (no S x S in HBM).
// GEMM uses double-buffered LDS fed by GLOBAL_LOAD_ASYNC_TO_LDS_B128
// (ASYNCcnt-tracked, no VGPR round trip).
// ---------------------------------------------------------------------------

#define D_MODEL 1024
#define SEQ     2048
#define NBATCH  2
#define NHEADS  16
#define DK      64

typedef __attribute__((ext_vector_type(16))) __bf16 v16bf;
typedef __attribute__((ext_vector_type(8)))  float  v8f;
typedef int v4i __attribute__((vector_size(16)));   // b128 payload type

union FragAB { v16bf v; unsigned short u[16]; };

#define AS1 __attribute__((address_space(1)))
#define AS3 __attribute__((address_space(3)))

#if defined(__has_builtin)
#if __has_builtin(__builtin_amdgcn_global_load_async_to_lds_b128)
#define HAVE_ASYNC_COPY 1
#endif
#endif

// 16B global -> LDS copy. Async path bypasses VGPRs, tracked by ASYNCcnt.
__device__ __forceinline__ void copy_b128_g2l(const unsigned short* g, unsigned short* l) {
#ifdef HAVE_ASYNC_COPY
    __builtin_amdgcn_global_load_async_to_lds_b128(
        (AS1 v4i*)(size_t)g, (AS3 v4i*)(size_t)l, 0, 0);
#else
    *(uint4*)l = *(const uint4*)g;
#endif
}

#ifdef HAVE_ASYNC_COPY
#if __has_builtin(__builtin_amdgcn_s_wait_asynccnt)
#define WAIT_ASYNC(n) __builtin_amdgcn_s_wait_asynccnt(n)
#else
#define WAIT_ASYNC(n) asm volatile("s_wait_asynccnt " #n)
#endif
#else
#define WAIT_ASYNC(n) do {} while (0)
#endif

// hardware f32 -> bf16 convert (single v_cvt, RNE)
__device__ __forceinline__ unsigned short f2bf(float f) {
    union { __bf16 b; unsigned short u; } c;
    c.b = (__bf16)f;
    return c.u;
}

// ---------------------------------------------------------------------------
// Kernel 0: fp32 -> bf16 conversion (vectorized, used for weights + q/k/v)
// ---------------------------------------------------------------------------
__global__ __launch_bounds__(256) void cvt_f32_bf16(const float* __restrict__ src,
                                                    unsigned short* __restrict__ dst,
                                                    int n4) {
    int i = blockIdx.x * blockDim.x + threadIdx.x;
    if (i < n4) {
        float4 v = ((const float4*)src)[i];
        unsigned p0 = (unsigned)f2bf(v.x) | ((unsigned)f2bf(v.y) << 16);
        unsigned p1 = (unsigned)f2bf(v.z) | ((unsigned)f2bf(v.w) << 16);
        ((uint2*)dst)[i] = make_uint2(p0, p1);
    }
}

// ---------------------------------------------------------------------------
// Kernel 1: GEMM  Y[M=4096,N=1024] = X(bf16) * W^T(bf16) + bias
//   mode 0/1: Q/K bf16 [B,H,S,DK];  mode 2: V^T bf16 [B,H,DK,S]
//   mode 3: fp32 row-major [B*S, D] -> d_out
// 256 thr (8 waves) -> 128x128 tile; wave -> 32x64; K-tile 64, double-buffered
// LDS with async global->LDS staging overlapped against 16 WMMA per stage.
// ---------------------------------------------------------------------------
__global__ __launch_bounds__(256) void gemm_bf16(const unsigned short* __restrict__ X,
                                                 const unsigned short* __restrict__ W,
                                                 const float* __restrict__ bias,
                                                 void* __restrict__ out,
                                                 int mode) {
    __shared__ unsigned short As[2][128 * 72];   // 128 rows x 64 K, pad->72
    __shared__ unsigned short Bs[2][128 * 72];

    const int tid  = threadIdx.x;
    const int lane = tid & 31;
    const int wave = tid >> 5;
    const int wy   = wave & 3;   // 4 waves along M
    const int wx   = wave >> 2;  // 2 waves along N
    const int blockM = blockIdx.y * 128;
    const int blockN = blockIdx.x * 128;
    const int ln   = lane & 15;
    const int half = lane >> 4;

    v8f acc[2][4];
#pragma unroll
    for (int mt = 0; mt < 2; mt++)
#pragma unroll
        for (int nt = 0; nt < 4; nt++)
            acc[mt][nt] = (v8f){0.f,0.f,0.f,0.f,0.f,0.f,0.f,0.f};

    const int rrow = tid >> 3;        // 0..31
    const int c8   = (tid & 7) * 8;   // 0..56 step 8 ushorts (16B)
    const unsigned short* xsrc = X + (size_t)(blockM + rrow) * D_MODEL + c8;
    const unsigned short* wsrc = W + (size_t)(blockN + rrow) * D_MODEL + c8;
    const int ldsoff = rrow * 72 + c8;

    // issue tile 0 into buffer 0 (8 async b128 per thread, in-order completion)
#pragma unroll
    for (int p = 0; p < 4; p++) {
        copy_b128_g2l(xsrc + (size_t)p * 32 * D_MODEL, &As[0][ldsoff + p * (32 * 72)]);
        copy_b128_g2l(wsrc + (size_t)p * 32 * D_MODEL, &Bs[0][ldsoff + p * (32 * 72)]);
    }

    for (int t = 0; t < D_MODEL / 64; ++t) {
        const int cur = t & 1;
        if (t + 1 < D_MODEL / 64) {
            const int k0n = (t + 1) * 64;   // stage next tile into other buffer
#pragma unroll
            for (int p = 0; p < 4; p++) {
                copy_b128_g2l(xsrc + (size_t)p * 32 * D_MODEL + k0n,
                              &As[1 - cur][ldsoff + p * (32 * 72)]);
                copy_b128_g2l(wsrc + (size_t)p * 32 * D_MODEL + k0n,
                              &Bs[1 - cur][ldsoff + p * (32 * 72)]);
            }
            WAIT_ASYNC(8);   // tile t's 8 ops done; tile t+1's 8 still in flight
        } else {
            WAIT_ASYNC(0);
        }
        __syncthreads();     // tile t visible to all waves

        const unsigned short* a0 = As[cur];
        const unsigned short* b0 = Bs[cur];
#pragma unroll
        for (int ks = 0; ks < 64; ks += 32) {
            FragAB af[2];
            const int kb = half * 8;
#pragma unroll
            for (int mt = 0; mt < 2; mt++) {
                const unsigned short* s = &a0[(wy * 32 + mt * 16 + ln) * 72 + ks];
#pragma unroll
                for (int i = 0; i < 8; i++) {
                    af[mt].u[i]     = s[kb + i];
                    af[mt].u[8 + i] = s[kb + 16 + i];
                }
            }
            const int kb2 = half * 8;
#pragma unroll
            for (int nt = 0; nt < 4; nt++) {
                FragAB bf_;
                const unsigned short* s = &b0[(wx * 64 + nt * 16 + ln) * 72 + ks + kb2];
#pragma unroll
                for (int i = 0; i < 8; i++) { bf_.u[i] = s[i]; bf_.u[8 + i] = s[16 + i]; }
#pragma unroll
                for (int mt = 0; mt < 2; mt++)
                    acc[mt][nt] = __builtin_amdgcn_wmma_f32_16x16x32_bf16(
                        false, af[mt].v, false, bf_.v, (short)0, acc[mt][nt], false, false);
            }
        }
        __syncthreads();     // all waves done reading buf[cur] before reuse
    }

    // epilogue: bias + scatter per mode
#pragma unroll
    for (int mt = 0; mt < 2; mt++) {
#pragma unroll
        for (int nt = 0; nt < 4; nt++) {
            int gc = blockN + wx * 64 + nt * 16 + ln;       // output feature e
            float bv = bias[gc];
#pragma unroll
            for (int j = 0; j < 8; j++) {
                int gr = blockM + wy * 32 + mt * 16 + half * 8 + j;  // b*S + s
                float val = acc[mt][nt][j] + bv;
                if (mode == 3) {
                    ((float*)out)[(size_t)gr * D_MODEL + gc] = val;
                } else {
                    int b = gr >> 11, s = gr & (SEQ - 1);
                    int h = gc >> 6,  d = gc & (DK - 1);
                    unsigned short o = f2bf(val);
                    if (mode == 2)  // V transposed: [B,H,DK,S]
                        ((unsigned short*)out)[(((size_t)b * NHEADS + h) * DK + d) * SEQ + s] = o;
                    else            // Q,K: [B,H,S,DK]
                        ((unsigned short*)out)[(((size_t)b * NHEADS + h) * SEQ + s) * DK + d] = o;
                }
            }
        }
    }
}

// ---------------------------------------------------------------------------
// Kernel 2: flash attention. Grid (S/64, B*H), 128 threads (4 waves).
// Each wave owns 16 query rows; streams 64-key blocks with online softmax.
// Output written directly in bf16 (feeds the final projection GEMM).
// ---------------------------------------------------------------------------
__global__ __launch_bounds__(128) void flash_attn(const unsigned short* __restrict__ Qp,
                                                  const unsigned short* __restrict__ Kp,
                                                  const unsigned short* __restrict__ Vt,
                                                  unsigned short* __restrict__ AT) {
    __shared__ unsigned short Plds[4][16][72];   // per-wave P tile, padded

    const int tid  = threadIdx.x;
    const int lane = tid & 31;
    const int wave = tid >> 5;
    const int ln   = lane & 15;
    const int half = lane >> 4;
    const int bh   = blockIdx.y;
    const int qrow = blockIdx.x * 64 + wave * 16;

    // Q fragments (persist whole kernel)
    FragAB qf[2];
    {
        const unsigned short* q = Qp + ((size_t)bh * SEQ + qrow + ln) * DK;
        const int kb = half * 8;
#pragma unroll
        for (int f = 0; f < 2; f++)
#pragma unroll
            for (int i = 0; i < 8; i++) {
                qf[f].u[i]     = q[f * 32 + kb + i];
                qf[f].u[8 + i] = q[f * 32 + kb + 16 + i];
            }
    }

    float m[8], l[8];
    v8f oacc[4];
#pragma unroll
    for (int j = 0; j < 8; j++) { m[j] = -1e30f; l[j] = 0.f; }
#pragma unroll
    for (int nt = 0; nt < 4; nt++) oacc[nt] = (v8f){0.f,0.f,0.f,0.f,0.f,0.f,0.f,0.f};

    const float SCL = 0.125f * 1.44269504088896f;   // (1/sqrt(dk)) * log2(e)

    for (int kb0 = 0; kb0 < SEQ; kb0 += 64) {
        if (kb0 + 64 < SEQ) {   // prefetch next K/V block -> global_prefetch_b8
            __builtin_prefetch(Kp + ((size_t)bh * SEQ + kb0 + 64 + lane * 2) * DK, 0, 1);
            __builtin_prefetch(Vt + ((size_t)bh * DK + lane * 2) * SEQ + kb0 + 64, 0, 1);
        }

        // scores: 16 x 64 tile = 4 WMMA accumulators
        v8f sacc[4];
#pragma unroll
        for (int nt = 0; nt < 4; nt++) {
            v8f s = (v8f){0.f,0.f,0.f,0.f,0.f,0.f,0.f,0.f};
#pragma unroll
            for (int f = 0; f < 2; f++) {
                FragAB kf;
                const unsigned short* kr =
                    Kp + ((size_t)bh * SEQ + kb0 + nt * 16 + ln) * DK + f * 32 + half * 8;
#pragma unroll
                for (int i = 0; i < 8; i++) { kf.u[i] = kr[i]; kf.u[8 + i] = kr[16 + i]; }
                s = __builtin_amdgcn_wmma_f32_16x16x32_bf16(
                        false, qf[f].v, false, kf.v, (short)0, s, false, false);
            }
            sacc[nt] = s;
        }

        // online softmax (rows live in VGPR slot j; reduce across 16-lane half)
#pragma unroll
        for (int j = 0; j < 8; j++) {
            float mx = fmaxf(fmaxf(sacc[0][j], sacc[1][j]), fmaxf(sacc[2][j], sacc[3][j]));
#pragma unroll
            for (int off = 1; off < 16; off <<= 1) mx = fmaxf(mx, __shfl_xor(mx, off, 32));
            float mnew  = fmaxf(m[j], mx);
            float alpha = exp2f((m[j] - mnew) * SCL);
            float rs = 0.f;
#pragma unroll
            for (int nt = 0; nt < 4; nt++) {
                float p = exp2f((sacc[nt][j] - mnew) * SCL);
                sacc[nt][j] = p;
                rs += p;
                oacc[nt][j] *= alpha;
            }
#pragma unroll
            for (int off = 1; off < 16; off <<= 1) rs += __shfl_xor(rs, off, 32);
            l[j] = l[j] * alpha + rs;
            m[j] = mnew;
        }

        // P (bf16) -> per-wave LDS region (C-layout -> A-layout re-striping)
#pragma unroll
        for (int nt = 0; nt < 4; nt++)
#pragma unroll
            for (int j = 0; j < 8; j++)
                Plds[wave][half * 8 + j][nt * 16 + ln] = f2bf(sacc[nt][j]);
        // same-wave DS ordering guarantees store->load; compiler emits s_wait_dscnt

        // O += P * V
#pragma unroll
        for (int f2 = 0; f2 < 2; f2++) {
            FragAB pf;
            const unsigned short* pr = &Plds[wave][ln][f2 * 32 + half * 8];
#pragma unroll
            for (int i = 0; i < 8; i++) { pf.u[i] = pr[i]; pf.u[8 + i] = pr[16 + i]; }
#pragma unroll
            for (int nt = 0; nt < 4; nt++) {
                FragAB vf;
                const unsigned short* vr =
                    Vt + ((size_t)bh * DK + nt * 16 + ln) * SEQ + kb0 + f2 * 32 + half * 8;
#pragma unroll
                for (int i = 0; i < 8; i++) { vf.u[i] = vr[i]; vf.u[8 + i] = vr[16 + i]; }
                oacc[nt] = __builtin_amdgcn_wmma_f32_16x16x32_bf16(
                               false, pf.v, false, vf.v, (short)0, oacc[nt], false, false);
            }
        }
    }

    // epilogue: normalize, write concat-heads bf16 [B,S,D]
    const int b = bh >> 4, h = bh & (NHEADS - 1);
#pragma unroll
    for (int j = 0; j < 8; j++) {
        float inv = 1.0f / l[j];
        int s = qrow + half * 8 + j;
        unsigned short* orow = AT + ((size_t)b * SEQ + s) * D_MODEL + h * DK;
#pragma unroll
        for (int nt = 0; nt < 4; nt++)
            orow[nt * 16 + ln] = f2bf(oacc[nt][j] * inv);
    }
}

// ---------------------------------------------------------------------------
// Host-side orchestration (all on `stream`; scratch in d_ws):
//   ws:  [0,8MB)    Wq,Wk,Wv,Wo bf16 (2MB each)
//        [8,16MB)   q  bf16    [B*S, D]
//        [16,24MB)  k  bf16    [B*S, D]
//        [24,32MB)  v  bf16    [B*S, D]
//        [32,40MB)  Q  bf16    [B,H,S,DK]
//        [40,48MB)  K  bf16    [B,H,S,DK]
//        [48,56MB)  V^T bf16   [B,H,DK,S]
//        [56,64MB)  attn out bf16 [B,S,D]
// ---------------------------------------------------------------------------
extern "C" void kernel_launch(void* const* d_in, const int* in_sizes, int n_in,
                              void* d_out, int out_size, void* d_ws, size_t ws_size,
                              hipStream_t stream) {
    const float* q  = (const float*)d_in[0];
    const float* k  = (const float*)d_in[1];
    const float* v  = (const float*)d_in[2];
    const float* Wq = (const float*)d_in[3];
    const float* bq = (const float*)d_in[4];
    const float* Wk = (const float*)d_in[5];
    const float* bk = (const float*)d_in[6];
    const float* Wv = (const float*)d_in[7];
    const float* bv = (const float*)d_in[8];
    const float* Wo = (const float*)d_in[9];
    const float* bo = (const float*)d_in[10];

    char* ws = (char*)d_ws;
    const size_t MB = 1024 * 1024;
    unsigned short* Wq_bf = (unsigned short*)(ws + 0 * MB);
    unsigned short* Wk_bf = (unsigned short*)(ws + 2 * MB);
    unsigned short* Wv_bf = (unsigned short*)(ws + 4 * MB);
    unsigned short* Wo_bf = (unsigned short*)(ws + 6 * MB);
    unsigned short* q_bf  = (unsigned short*)(ws + 8 * MB);
    unsigned short* k_bf  = (unsigned short*)(ws + 16 * MB);
    unsigned short* v_bf  = (unsigned short*)(ws + 24 * MB);
    unsigned short* Qp    = (unsigned short*)(ws + 32 * MB);
    unsigned short* Kp    = (unsigned short*)(ws + 40 * MB);
    unsigned short* Vt    = (unsigned short*)(ws + 48 * MB);
    unsigned short* AT    = (unsigned short*)(ws + 56 * MB);

    const int nW4 = (D_MODEL * D_MODEL) / 4;            // float4 per weight
    const int nX4 = (NBATCH * SEQ * D_MODEL) / 4;       // float4 per activation
    cvt_f32_bf16<<<(nW4 + 255) / 256, 256, 0, stream>>>(Wq, Wq_bf, nW4);
    cvt_f32_bf16<<<(nW4 + 255) / 256, 256, 0, stream>>>(Wk, Wk_bf, nW4);
    cvt_f32_bf16<<<(nW4 + 255) / 256, 256, 0, stream>>>(Wv, Wv_bf, nW4);
    cvt_f32_bf16<<<(nW4 + 255) / 256, 256, 0, stream>>>(Wo, Wo_bf, nW4);
    cvt_f32_bf16<<<(nX4 + 255) / 256, 256, 0, stream>>>(q, q_bf, nX4);
    cvt_f32_bf16<<<(nX4 + 255) / 256, 256, 0, stream>>>(k, k_bf, nX4);
    cvt_f32_bf16<<<(nX4 + 255) / 256, 256, 0, stream>>>(v, v_bf, nX4);

    dim3 ggrid(D_MODEL / 128, (NBATCH * SEQ) / 128);    // (8, 32)
    gemm_bf16<<<ggrid, 256, 0, stream>>>(q_bf, Wq_bf, bq, (void*)Qp, 0);
    gemm_bf16<<<ggrid, 256, 0, stream>>>(k_bf, Wk_bf, bk, (void*)Kp, 1);
    gemm_bf16<<<ggrid, 256, 0, stream>>>(v_bf, Wv_bf, bv, (void*)Vt, 2);

    dim3 fgrid(SEQ / 64, NBATCH * NHEADS);              // (32, 32)
    flash_attn<<<fgrid, 128, 0, stream>>>(Qp, Kp, Vt, AT);

    gemm_bf16<<<ggrid, 256, 0, stream>>>(AT, Wo_bf, bo, d_out, 3);
}
